// GCN_12824772346523
// MI455X (gfx1250) — matrix operand order repaired and checked by reading, not compiled, not run on hardware
//
#include <hip/hip_runtime.h>

typedef __bf16 bf16;
typedef __attribute__((ext_vector_type(16))) __bf16 v16bf;
typedef __attribute__((ext_vector_type(8)))  __bf16 v8bf;
typedef __attribute__((ext_vector_type(8)))  float  v8f;

static __device__ __forceinline__ bf16 f2bf(float f) { return (bf16)f; }

static __device__ __forceinline__ void atomic_fadd(float* p, float v) {
  __hip_atomic_fetch_add(p, v, __ATOMIC_RELAXED, __HIP_MEMORY_SCOPE_AGENT);
}

// ---------------------------------------------------------------------------
// Aggregation: h = x (copy), then h[dst] += x[src] per edge (L2-resident atomics)
// ---------------------------------------------------------------------------
__global__ __launch_bounds__(256) void gin_copy_f4(const float4* __restrict__ in,
                                                   float4* __restrict__ out, int n4) {
  int i = blockIdx.x * 256 + threadIdx.x;
  if (i < n4) out[i] = in[i];
}

__global__ __launch_bounds__(256) void gin_scatter128(const float* __restrict__ x,
                                                      const int* __restrict__ src,
                                                      const int* __restrict__ dst,
                                                      float* __restrict__ h, int E) {
  int e = blockIdx.x * 8 + (threadIdx.x >> 5);
  if (e >= E) return;
  int lane = threadIdx.x & 31;
  int s = src[e];
  int d = dst[e];
  float4 v = *(const float4*)(x + (size_t)s * 128 + lane * 4);
  float* p = h + (size_t)d * 128 + lane * 4;
  atomic_fadd(p + 0, v.x);
  atomic_fadd(p + 1, v.y);
  atomic_fadd(p + 2, v.z);
  atomic_fadd(p + 3, v.w);
}

// ---------------------------------------------------------------------------
// WMMA GEMM: out[N x K2] = act(in[N x K1] @ W[K1 x K2] + bias)
// bf16 operands, f32 accumulate via v_wmma_f32_16x16x32_bf16.
// Weights staged into LDS pre-swizzled into B-fragment order:
//   wlds[((nt*KS+ks)*32 + lane)*16 + i]  <=>  W[ks*32 + (i&7) + 16*(i>=8) + 8*(lane>=16)][nt*16 + (lane&15)]
// so each lane's fragment is one contiguous 32-byte LDS read.
// ---------------------------------------------------------------------------
template <int K1, int K2, bool INB, bool RELU, bool OUTB>
__global__ __launch_bounds__(256) void gemm_wmma(const void* __restrict__ in_,
                                                 const float* __restrict__ w,
                                                 const float* __restrict__ bias,
                                                 void* __restrict__ out_, int Nrows) {
  constexpr int KS = K1 / 32;          // K-steps of 32
  constexpr int NT = (K2 + 15) / 16;   // N tiles of 16
  __shared__ bf16 wlds[NT * KS * 512];

  for (int j = threadIdx.x; j < NT * KS * 512; j += 256) {
    int i    = j & 15;
    int lane = (j >> 4) & 31;
    int blk  = j >> 9;
    int ks   = blk % KS;
    int nt   = blk / KS;
    int k    = ks * 32 + (i & 7) + ((i & 8) ? 16 : 0) + ((lane & 16) ? 8 : 0);
    int n    = nt * 16 + (lane & 15);
    wlds[j]  = f2bf((n < K2) ? w[k * K2 + n] : 0.0f);
  }
  __syncthreads();

  int lane   = threadIdx.x & 31;
  int wid    = threadIdx.x >> 5;
  int tile_m = blockIdx.x * 128 + wid * 16;
  int row    = tile_m + (lane & 15);
  if (row >= Nrows) row = Nrows - 1;   // clamp (stores are guarded)
  int khalf = (lane & 16) ? 8 : 0;

  // A fragments: row M = lane&15; lane-half selects K {0..7,16..23} vs {8..15,24..31}
  v16bf afrag[KS];
  if constexpr (INB) {
    const bf16* rp = (const bf16*)in_ + (size_t)row * K1;
#pragma unroll
    for (int ks = 0; ks < KS; ++ks) {
      int koff = ks * 32 + khalf;
      v8bf lo = *(const v8bf*)(rp + koff);
      v8bf hi = *(const v8bf*)(rp + koff + 16);
      v16bf a;
#pragma unroll
      for (int i = 0; i < 8; ++i) { a[i] = lo[i]; a[i + 8] = hi[i]; }
      afrag[ks] = a;
    }
  } else {
    const float* rp = (const float*)in_ + (size_t)row * K1;
#pragma unroll
    for (int ks = 0; ks < KS; ++ks) {
      int koff = ks * 32 + khalf;
      float4 f0 = *(const float4*)(rp + koff);
      float4 f1 = *(const float4*)(rp + koff + 4);
      float4 f2 = *(const float4*)(rp + koff + 16);
      float4 f3 = *(const float4*)(rp + koff + 20);
      v16bf a;
      a[0]  = f2bf(f0.x); a[1]  = f2bf(f0.y); a[2]  = f2bf(f0.z); a[3]  = f2bf(f0.w);
      a[4]  = f2bf(f1.x); a[5]  = f2bf(f1.y); a[6]  = f2bf(f1.z); a[7]  = f2bf(f1.w);
      a[8]  = f2bf(f2.x); a[9]  = f2bf(f2.y); a[10] = f2bf(f2.z); a[11] = f2bf(f2.w);
      a[12] = f2bf(f3.x); a[13] = f2bf(f3.y); a[14] = f2bf(f3.z); a[15] = f2bf(f3.w);
      afrag[ks] = a;
    }
  }

  int col_lane = lane & 15;
  int rbase    = tile_m + ((lane & 16) ? 8 : 0);
#pragma unroll
  for (int nt = 0; nt < NT; ++nt) {
    v8f acc = {0.f, 0.f, 0.f, 0.f, 0.f, 0.f, 0.f, 0.f};
#pragma unroll
    for (int ks = 0; ks < KS; ++ks) {
      v16bf b = *(const v16bf*)&wlds[((nt * KS + ks) << 9) + (lane << 4)];
      acc = __builtin_amdgcn_wmma_f32_16x16x32_bf16(false, afrag[ks], false, b,
                                                    (short)0, acc, false, false);
    }
    int col = nt * 16 + col_lane;
    if (col < K2) {
      float bv = bias[col];
#pragma unroll
      for (int r = 0; r < 8; ++r) {
        int orow = rbase + r;
        if (orow < Nrows) {
          float v = acc[r] + bv;
          if constexpr (RELU) v = fmaxf(v, 0.0f);
          if constexpr (OUTB) ((bf16*)out_)[(size_t)orow * K2 + col] = f2bf(v);
          else                ((float*)out_)[(size_t)orow * K2 + col] = v;
        }
      }
    }
  }
}

// ---------------------------------------------------------------------------
// BatchNorm (training mode, biased var) + ReLU + optional residual add
// ---------------------------------------------------------------------------
__global__ __launch_bounds__(256) void bn_zero(float* p) { p[threadIdx.x] = 0.0f; }

__global__ __launch_bounds__(128) void bn_stats(const float* __restrict__ y,
                                                float* __restrict__ stats,
                                                int N, int rows) {
  int c  = threadIdx.x;               // channel 0..127
  int r0 = blockIdx.x * rows;
  int r1 = r0 + rows; if (r1 > N) r1 = N;
  float s = 0.0f, s2 = 0.0f;
  for (int r = r0; r < r1; ++r) {
    float v = y[(size_t)r * 128 + c];
    s += v; s2 += v * v;
  }
  atomic_fadd(&stats[c], s);
  atomic_fadd(&stats[128 + c], s2);
}

__global__ __launch_bounds__(256) void bn_apply(const float* __restrict__ y,
                                                const float* __restrict__ stats,
                                                const float* __restrict__ g,
                                                const float* __restrict__ beta,
                                                const float* __restrict__ resid,
                                                float* __restrict__ out,
                                                int n, float invN) {
  int i = blockIdx.x * 256 + threadIdx.x;
  if (i >= n) return;
  int c = i & 127;
  float m   = stats[c] * invN;
  float var = stats[128 + c] * invN - m * m;
  if (var < 0.0f) var = 0.0f;
  float rs = rsqrtf(var + 1e-5f);
  float v  = fmaxf((y[i] - m) * rs * g[c] + beta[c], 0.0f);
  if (resid) v += resid[i];
  out[i] = v;
}

// ---------------------------------------------------------------------------
extern "C" void kernel_launch(void* const* d_in, const int* in_sizes, int n_in,
                              void* d_out, int out_size, void* d_ws, size_t ws_size,
                              hipStream_t stream) {
  const float* x   = (const float*)d_in[0];
  const int*   ei  = (const int*)d_in[1];
  const int N = in_sizes[0] / 128;
  const int E = in_sizes[1] / 2;
  const int* src = ei;
  const int* dst = ei + E;

  const float* head_w1   = (const float*)d_in[2];
  const float* head_b1   = (const float*)d_in[3];
  const float* head_w2   = (const float*)d_in[4];
  const float* head_b2   = (const float*)d_in[5];
  const float* head_g    = (const float*)d_in[6];
  const float* head_beta = (const float*)d_in[7];
  const float* res_w1    = (const float*)d_in[8];
  const float* res_b1    = (const float*)d_in[9];
  const float* res_w2    = (const float*)d_in[10];
  const float* res_b2    = (const float*)d_in[11];
  const float* res_g     = (const float*)d_in[12];
  const float* res_beta  = (const float*)d_in[13];
  const float* tail_w1   = (const float*)d_in[14];
  const float* tail_b1   = (const float*)d_in[15];
  const float* tail_w2   = (const float*)d_in[16];
  const float* tail_b2   = (const float*)d_in[17];

  const size_t nb = (size_t)N * 128 * sizeof(float);   // 51.2 MB
  char* ws = (char*)d_ws;
  float* P0    = (float*)ws;               // feature ping (identity)
  float* P1    = (float*)(ws + nb);        // feature pong
  float* H     = (float*)(ws + 2 * nb);    // agg buffer, reused as GEMM2 output
  bf16*  T     = (bf16*)(ws + 3 * nb);     // MLP hidden (bf16, up to N*256)
  float* stats = (float*)(ws + 4 * nb);    // 256 floats: sum / sumsq

  const int n4    = N * 32;                // N*128/4
  const int nelem = N * 128;
  const float invN = 1.0f / (float)N;
  const int rpb = 400;

  dim3 cg((n4 + 255) / 256);
  dim3 sg((E + 7) / 8);
  dim3 gg((N + 127) / 128);
  dim3 ag((nelem + 255) / 256);
  dim3 stg((N + rpb - 1) / rpb);

  auto agg = [&](const float* xin) {
    gin_copy_f4<<<cg, 256, 0, stream>>>((const float4*)xin, (float4*)H, n4);
    gin_scatter128<<<sg, 256, 0, stream>>>(xin, src, dst, H, E);
  };
  auto bn = [&](const float* g, const float* b, const float* resid, float* out) {
    bn_zero<<<1, 256, 0, stream>>>(stats);
    bn_stats<<<stg, 128, 0, stream>>>(H, stats, N, rpb);
    bn_apply<<<ag, 256, 0, stream>>>(H, stats, g, b, resid, out, nelem, invN);
  };

  // ---- head: GIN(128->256->128) -> BN -> ReLU ----
  agg(x);
  gemm_wmma<128, 256, false, true,  true ><<<gg, 256, 0, stream>>>(H, head_w1, head_b1, T, N);
  gemm_wmma<256, 128, true,  false, false><<<gg, 256, 0, stream>>>(T, head_w2, head_b2, H, N);
  bn(head_g, head_beta, nullptr, P0);

  // ---- 3 residual blocks, 2 units each ----
  for (int blk = 0; blk < 3; ++blk) {
    int u0 = 2 * blk, u1 = 2 * blk + 1;
    // unit u0: P0 -> P1
    agg(P0);
    gemm_wmma<128, 128, false, true,  true ><<<gg, 256, 0, stream>>>(
        H, res_w1 + (size_t)u0 * 128 * 128, res_b1 + u0 * 128, T, N);
    gemm_wmma<128, 128, true,  false, false><<<gg, 256, 0, stream>>>(
        T, res_w2 + (size_t)u0 * 128 * 128, res_b2 + u0 * 128, H, N);
    bn(res_g + u0 * 128, res_beta + u0 * 128, nullptr, P1);
    // unit u1: P1 -> P0, fused residual add of identity (P0, in place)
    agg(P1);
    gemm_wmma<128, 128, false, true,  true ><<<gg, 256, 0, stream>>>(
        H, res_w1 + (size_t)u1 * 128 * 128, res_b1 + u1 * 128, T, N);
    gemm_wmma<128, 128, true,  false, false><<<gg, 256, 0, stream>>>(
        T, res_w2 + (size_t)u1 * 128 * 128, res_b2 + u1 * 128, H, N);
    bn(res_g + u1 * 128, res_beta + u1 * 128, P0, P0);
  }

  // ---- tail: GIN(128->32->3), no BN/ReLU on output ----
  agg(P0);
  gemm_wmma<128, 32, false, true,  true ><<<gg, 256, 0, stream>>>(H, tail_w1, tail_b1, T, N);
  gemm_wmma< 32,  3, true,  false, false><<<gg, 256, 0, stream>>>(T, tail_w2, tail_b2, d_out, N);
}
